// LongContextAttentionQKVPacked_42228118455017
// MI455X (gfx1250) — compile-verified, matrix operand order checked by reading, not run on hardware
//
#include <hip/hip_runtime.h>

typedef __attribute__((ext_vector_type(16))) _Float16 v16h;
typedef __attribute__((ext_vector_type(4)))  _Float16 v4h;
typedef __attribute__((ext_vector_type(8)))  float    v8f;
typedef __attribute__((ext_vector_type(4)))  float    v4f;

#define B_  2
#define S_  2048
#define H_  32
#define D_  128
#define QT  64     // query rows per workgroup (4 waves x 16)
#define KT  64     // keys per chunk (4 x 16-key subtiles)

#if __has_builtin(__builtin_amdgcn_exp2f)
#define EXP2F(x) __builtin_amdgcn_exp2f(x)
#else
#define EXP2F(x) exp2f(x)
#endif
#if __has_builtin(__builtin_amdgcn_rcpf)
#define RCPF(x) __builtin_amdgcn_rcpf(x)
#else
#define RCPF(x) (1.0f / (x))
#endif

__global__ __launch_bounds__(128)
void fa_fwd_wmma_gfx1250(const float* __restrict__ qkv,
                         const int* __restrict__ causal_p,
                         float* __restrict__ out)
{
    // double-buffered K (row major) and V (transposed), f16: 4 x 16KB = 64KB
    __shared__ _Float16 Ks[2][KT * D_];   // [key][d]
    __shared__ _Float16 Vt[2][D_ * KT];   // [d][key]

    const int lane = threadIdx.x & 31;
    const int wave = threadIdx.x >> 5;
    const int ln   = lane & 15;          // query / N index within tile
    const int kbs  = (lane >> 4) * 8;    // K/M base for this half-wave

    const int qt0 = blockIdx.x * QT;
    const int bh  = blockIdx.y;
    const int b   = bh / H_;
    const int h   = bh % H_;
    const int causal = causal_p[0];

    const int    rowQ  = qt0 + wave * 16 + ln;    // this lane's query row
    const int    qMinW = qt0 + wave * 16;         // wave's lowest query row
    const size_t srow  = (size_t)3 * H_ * D_;     // floats per sequence step

    // ---- Q fragments (B-operand layout for Q^T) ----
    // pre-scaled by log2(e)/sqrt(D): logits come out in exp2 domain, so the
    // softmax needs only v_sub + v_exp_f32 per element (no log2e muls).
    const float  qscale = 0.08838834764831845f * 1.4426950408889634f;
    const float* Qrow   = qkv + (size_t)(b * S_ + rowQ) * srow + h * D_;
    v16h qf[4];
#pragma unroll
    for (int sl = 0; sl < 4; ++sl) {
        const int dd0 = sl * 32;
#pragma unroll
        for (int j = 0; j < 8; ++j) {
            qf[sl][j]     = (_Float16)(Qrow[dd0 + kbs + j] * qscale);
            qf[sl][8 + j] = (_Float16)(Qrow[dd0 + 16 + kbs + j] * qscale);
        }
    }

    // ---- cooperative chunk staging: f32 -> f16, V transposed, float4 loads ----
    auto stage = [&](int kb, int buf) {
#pragma unroll
        for (int i = 0; i < 16; ++i) {
            const int idx = threadIdx.x + i * 128;     // 2048 float4 per matrix
            const int ky  = idx >> 5;                  // 32 float4 per row
            const int d4  = (idx & 31) << 2;
            const float* gp = qkv + (size_t)(b * S_ + kb + ky) * srow + h * D_ + d4;
            const v4f kv = *(const v4f*)(gp + H_ * D_);
            const v4f vv = *(const v4f*)(gp + 2 * H_ * D_);
            v4h kh;
#pragma unroll
            for (int t = 0; t < 4; ++t) kh[t] = (_Float16)kv[t];
            *(v4h*)&Ks[buf][ky * D_ + d4] = kh;        // packed 8B ds_store
#pragma unroll
            for (int t = 0; t < 4; ++t)
                Vt[buf][(d4 + t) * KT + ky] = (_Float16)vv[t];
        }
    };

    // ---- running state ----
    v8f O[8];
#pragma unroll
    for (int n = 0; n < 8; ++n) O[n] = (v8f){};
    float Mrow = -1e30f;                 // log2-domain running max
    float Lrow = 0.0f;

    const int nChunks = causal ? (qt0 / KT + 1) : (S_ / KT);

    stage(0, 0);

    for (int c = 0; c < nChunks; ++c) {
        const int kb = c * KT;
        __syncthreads();                       // chunk c staged; buf (c+1)&1 free
        if (c + 1 < nChunks) stage((c + 1) * KT, (c + 1) & 1);
        if (c + 2 < nChunks) {                 // light prefetch of chunk c+2
            const int ky = threadIdx.x >> 1;
            const int dd = (threadIdx.x & 1) * 64;
            const float* gp = qkv + (size_t)(b * S_ + kb + 2 * KT + ky) * srow + h * D_ + dd;
            __builtin_prefetch(gp + H_ * D_, 0, 0);
            __builtin_prefetch(gp + 2 * H_ * D_, 0, 0);
        }

        const _Float16* ks = Ks[c & 1];
        const _Float16* vt = Vt[c & 1];

        // ---- S^T = K_tile x Q^T : four 16-key subtiles ----
        float sc[32];
#pragma unroll
        for (int sub = 0; sub < 4; ++sub) {
            const bool active = !(causal && (kb + sub * 16 > qMinW + 15));
            if (active) {
                const _Float16* kr = &ks[(sub * 16 + ln) * D_];
                v8f acc = (v8f){};
#pragma unroll
                for (int sl = 0; sl < 4; ++sl) {
                    const int dd0 = sl * 32;
                    v16h kf;
#pragma unroll
                    for (int j = 0; j < 8; ++j) {
                        kf[j]     = kr[dd0 + kbs + j];
                        kf[8 + j] = kr[dd0 + 16 + kbs + j];
                    }
                    acc = __builtin_amdgcn_wmma_f32_16x16x32_f16(
                              false, kf, false, qf[sl], (short)0, acc, false, false);
                }
                // mask only needed when this subtile straddles the diagonal
                const bool needMask = causal && (kb + sub * 16 + 15 > qMinW);
                if (needMask) {
#pragma unroll
                    for (int r = 0; r < 8; ++r) {
                        const int kg = kb + sub * 16 + kbs + r;
                        sc[sub * 8 + r] = (kg > rowQ) ? -1e30f : acc[r];
                    }
                } else {
#pragma unroll
                    for (int r = 0; r < 8; ++r) sc[sub * 8 + r] = acc[r];
                }
            } else {
#pragma unroll
                for (int r = 0; r < 8; ++r) sc[sub * 8 + r] = -1e30f;
            }
        }

        // ---- online softmax in exp2 domain (lanes ln/ln+16 share a row) ----
        float mloc = sc[0];
#pragma unroll
        for (int i = 1; i < 32; ++i) mloc = fmaxf(mloc, sc[i]);
        mloc = fmaxf(mloc, __shfl_xor(mloc, 16, 32));
        const float Mnew = fmaxf(Mrow, mloc);
        const float corr = EXP2F(Mrow - Mnew);

        float psum = 0.0f;
        v16h  pf0, pf1;
#pragma unroll
        for (int i = 0; i < 16; ++i) {
            const float p = EXP2F(sc[i] - Mnew);
            psum += p;
            pf0[i] = (_Float16)p;              // S^T C-layout == P A-layout
        }
#pragma unroll
        for (int i = 0; i < 16; ++i) {
            const float p = EXP2F(sc[16 + i] - Mnew);
            psum += p;
            pf1[i] = (_Float16)p;
        }
        psum += __shfl_xor(psum, 16, 32);
        Lrow = Lrow * corr + psum;
        Mrow = Mnew;

        // rescale O only when the running max actually moved (corr < 1)
        if (__any(corr < 1.0f)) {
            float fac[8];
#pragma unroll
            for (int r = 0; r < 8; ++r) fac[r] = __shfl(corr, r + kbs, 32);
#pragma unroll
            for (int n = 0; n < 8; ++n)
#pragma unroll
                for (int r = 0; r < 8; ++r) O[n][r] *= fac[r];
        }

        // ---- O += P x V : 8 column tiles of 16 over D, 2 key halves ----
#pragma unroll
        for (int n = 0; n < 8; ++n) {
            const _Float16* vr = &vt[(n * 16 + ln) * KT];
            v16h vf0, vf1;
#pragma unroll
            for (int j = 0; j < 8; ++j) {
                vf0[j]     = vr[kbs + j];
                vf0[8 + j] = vr[16 + kbs + j];
                vf1[j]     = vr[32 + kbs + j];
                vf1[8 + j] = vr[48 + kbs + j];
            }
            O[n] = __builtin_amdgcn_wmma_f32_16x16x32_f16(
                       false, pf0, false, vf0, (short)0, O[n], false, false);
            O[n] = __builtin_amdgcn_wmma_f32_16x16x32_f16(
                       false, pf1, false, vf1, (short)0, O[n], false, false);
        }
    }

    // ---- epilogue: normalize by row sum, store fp32 ----
    const float linv = RCPF(Lrow);
    float lf[8];
#pragma unroll
    for (int r = 0; r < 8; ++r) lf[r] = __shfl(linv, r + kbs, 32);

#pragma unroll
    for (int n = 0; n < 8; ++n) {
        const int col = n * 16 + ln;
#pragma unroll
        for (int r = 0; r < 8; ++r) {
            const int row = qt0 + wave * 16 + r + kbs;
            out[((size_t)(b * S_ + row) * H_ + h) * D_ + col] = O[n][r] * lf[r];
        }
    }
}

extern "C" void kernel_launch(void* const* d_in, const int* in_sizes, int n_in,
                              void* d_out, int out_size, void* d_ws, size_t ws_size,
                              hipStream_t stream) {
    (void)in_sizes; (void)n_in; (void)out_size; (void)d_ws; (void)ws_size;
    const float* qkv    = (const float*)d_in[0];
    const int*   causal = (const int*)d_in[1];
    float*       out    = (float*)d_out;

    dim3 grid(S_ / QT, B_ * H_);
    dim3 block(128);
    fa_fwd_wmma_gfx1250<<<grid, block, 0, stream>>>(qkv, causal, out);
}
